// CXLoss_953482740069
// MI455X (gfx1250) — matrix operand order repaired and checked by reading, not compiled, not run on hardware
//
#include <hip/hip_runtime.h>
#include <hip/hip_bf16.h>
#include <math.h>

// Problem constants (from reference setup_inputs)
#define N_   4
#define C_   256
#define P_   4096            // H*W = 64*64
#define EPS_ 1e-8f
#define INV_SIG (1.0f / (0.1f + 1e-8f))   // 1/(SIGMA+EPS)

typedef __attribute__((ext_vector_type(16))) __bf16 v16bf;
typedef __attribute__((ext_vector_type(8)))  __bf16 v8bf;
typedef __attribute__((ext_vector_type(8)))  float  v8f;

// ---------------------------------------------------------------------------
// Fragment loaders. Normalized features are stored bf16 as [n, p, c] row-major
// (row stride C_ = 256 elements = 512 B, so every load below is 16B-aligned).
//
// A (16x32, M x K) lane layout (CDNA5 ISA 7.12.2, 16-bit A):
//   lane L (0..15):  row m0+L,     elems 0..7 -> K = k0+0..7,   8..15 -> K = k0+16..23
//   lane L (16..31): row m0+L-16,  elems 0..7 -> K = k0+8..15,  8..15 -> K = k0+24..31
// => two contiguous 8-element (16 B) chunks at k0 + 8*(L>>4) and +16 more.
__device__ __forceinline__ v16bf load_afrag(const __bf16* __restrict__ fTt,
                                            int n, int m0, int k0, int lane) {
    int m = m0 + (lane & 15);
    int kb = k0 + 8 * (lane >> 4);
    const __bf16* row = fTt + (size_t)(n * P_ + m) * C_;
    v8bf lo = *(const v8bf*)(row + kb);
    v8bf hi = *(const v8bf*)(row + kb + 16);
    return __builtin_shufflevector(lo, hi, 0,1,2,3,4,5,6,7,8,9,10,11,12,13,14,15);
}

// B (32x16, K x N) lane layout matches A: lanes 0..15 hold K=k0+0..7,16..23 for N=lane,
// lanes 16..31 hold K=k0+8..15,24..31 for N=lane-16.
__device__ __forceinline__ v16bf load_bfrag(const __bf16* __restrict__ fIt,
                                            int n, int j0, int k0, int lane) {
    int j = j0 + (lane & 15);
    int kb = k0 + 8 * (lane >> 4);
    const __bf16* row = fIt + (size_t)(n * P_ + j) * C_;
    v8bf lo = *(const v8bf*)(row + kb);
    v8bf hi = *(const v8bf*)(row + kb + 16);
    return __builtin_shufflevector(lo, hi, 0,1,2,3,4,5,6,7,8,9,10,11,12,13,14,15);
}

// One 16x16 sim tile: K-loop over C_=256 in steps of 32 (8 WMMAs), B preloaded.
__device__ __forceinline__ v8f sim_tile(const __bf16* __restrict__ fTt,
                                        const v16bf* bfr, int n, int m0, int lane) {
    v8f acc = {};
    #pragma unroll
    for (int kk = 0; kk < 8; ++kk) {
        v16bf a = load_afrag(fTt, n, m0, kk * 32, lane);
        acc = __builtin_amdgcn_wmma_f32_16x16x32_bf16(
                  false, a, false, bfr[kk], (short)0, acc, false, false);
    }
    return acc;
}

// ---------------------------------------------------------------------------
// Kernel 1: per-channel mean of featureT over (N, H, W). One block per channel.
__global__ void cx_mean(const float* __restrict__ fT, float* __restrict__ meanT) {
    int c = blockIdx.x;
    __shared__ float sb[256];
    float s = 0.f;
    for (int i = threadIdx.x; i < N_ * P_; i += 256) {
        int n = i >> 12, hw = i & (P_ - 1);
        s += fT[(size_t)(n * C_ + c) * P_ + hw];
    }
    sb[threadIdx.x] = s;
    __syncthreads();
    for (int off = 128; off > 0; off >>= 1) {
        if (threadIdx.x < off) sb[threadIdx.x] += sb[threadIdx.x + off];
        __syncthreads();
    }
    if (threadIdx.x == 0) meanT[c] = sb[0] * (1.0f / (N_ * P_));
}

// Kernel 2: center by meanT, channel-wise L2 normalize, transpose to
// bf16 [n, p, c]. One thread per (n, hw); channel loads coalesce across lanes.
__global__ void cx_norm(const float* __restrict__ in, const float* __restrict__ meanT,
                        __bf16* __restrict__ outT) {
    int t = blockIdx.x * 256 + threadIdx.x;       // over N_*P_
    int n = t >> 12, hw = t & (P_ - 1);
    const float* base = in + (size_t)n * C_ * P_ + hw;
    float ss = 0.f;
    for (int c = 0; c < C_; ++c) {
        float x = base[(size_t)c * P_] - meanT[c];
        ss += x * x;
    }
    float inv = 1.0f / (sqrtf(ss) + EPS_);
    __bf16* orow = outT + (size_t)t * C_;
    for (int c = 0; c < C_; ++c) {
        float x = (base[(size_t)c * P_] - meanT[c]) * inv;
        orow[c] = (__bf16)x;
    }
}

// Kernel 3: column statistics. Block = 4 waves owns one 16-column tile and
// sweeps all 4096 rows twice: (1) column max(sim) -> min raw_dist, (2) exp-sum.
__global__ __launch_bounds__(128) void cx_colstats(const __bf16* __restrict__ fTt,
                                                   const __bf16* __restrict__ fIt,
                                                   float* __restrict__ colMin,
                                                   float* __restrict__ colS) {
    int n    = blockIdx.x >> 8;        // P_/16 = 256 column tiles per batch
    int j0   = (blockIdx.x & 255) * 16;
    int wave = threadIdx.x >> 5;
    int lane = threadIdx.x & 31;

    v16bf bfr[8];
    #pragma unroll
    for (int kk = 0; kk < 8; ++kk) bfr[kk] = load_bfrag(fIt, n, j0, kk * 32, lane);

    __shared__ float s_red[4][16];
    __shared__ float s_m[16];

    // Pass 1: per-column max of sim over all rows handled by this wave.
    float cmax = -1e30f;
    for (int mt = wave; mt < 256; mt += 4) {
        v8f acc = sim_tile(fTt, bfr, n, mt * 16, lane);
        #pragma unroll
        for (int r = 0; r < 8; ++r) cmax = fmaxf(cmax, acc[r]);
    }
    cmax = fmaxf(cmax, __shfl_xor(cmax, 16, 32));   // combine M-halves (same column)
    if (lane < 16) s_red[wave][lane] = cmax;
    __syncthreads();
    if (threadIdx.x < 16) {
        float m = fmaxf(fmaxf(s_red[0][threadIdx.x], s_red[1][threadIdx.x]),
                        fmaxf(s_red[2][threadIdx.x], s_red[3][threadIdx.x]));
        s_m[threadIdx.x] = (1.0f - m) * 0.5f;       // min over rows of raw_dist
    }
    __syncthreads();
    float invM = 1.0f / (s_m[lane & 15] + EPS_);

    // Pass 2: per-column sum of exp((1 - raw/min)/sigma).
    float ssum = 0.f;
    for (int mt = wave; mt < 256; mt += 4) {
        v8f acc = sim_tile(fTt, bfr, n, mt * 16, lane);
        #pragma unroll
        for (int r = 0; r < 8; ++r) {
            float raw = (1.0f - acc[r]) * 0.5f;
            ssum += __expf((1.0f - raw * invM) * INV_SIG);
        }
    }
    ssum += __shfl_xor(ssum, 16, 32);
    if (lane < 16) s_red[wave][lane] = ssum;
    __syncthreads();
    if (threadIdx.x < 16) {
        float s = s_red[0][threadIdx.x] + s_red[1][threadIdx.x]
                + s_red[2][threadIdx.x] + s_red[3][threadIdx.x];
        int j = n * P_ + j0 + threadIdx.x;
        colS[j]   = s;
        colMin[j] = s_m[threadIdx.x];
    }
}

// Kernel 4: per-row max of CX = W/(Wsum+eps). Block owns one 16-row tile,
// sweeps all columns; A fragments are loop-invariant and preloaded.
__global__ __launch_bounds__(128) void cx_rowmax(const __bf16* __restrict__ fTt,
                                                 const __bf16* __restrict__ fIt,
                                                 const float* __restrict__ colMin,
                                                 const float* __restrict__ colS,
                                                 float* __restrict__ rowMax) {
    int n    = blockIdx.x >> 8;
    int m0   = (blockIdx.x & 255) * 16;
    int wave = threadIdx.x >> 5;
    int lane = threadIdx.x & 31;

    v16bf afr[8];
    #pragma unroll
    for (int kk = 0; kk < 8; ++kk) afr[kk] = load_afrag(fTt, n, m0, kk * 32, lane);

    float rmax[8];
    #pragma unroll
    for (int r = 0; r < 8; ++r) rmax[r] = 0.f;      // CX >= 0

    for (int jt = wave; jt < 256; jt += 4) {
        int j0 = jt * 16;
        v8f acc = {};
        #pragma unroll
        for (int kk = 0; kk < 8; ++kk) {
            v16bf b = load_bfrag(fIt, n, j0, kk * 32, lane);
            acc = __builtin_amdgcn_wmma_f32_16x16x32_bf16(
                      false, afr[kk], false, b, (short)0, acc, false, false);
        }
        int j = n * P_ + j0 + (lane & 15);
        float invM = 1.0f / (colMin[j] + EPS_);
        float invS = 1.0f / (colS[j] + EPS_);
        #pragma unroll
        for (int r = 0; r < 8; ++r) {
            float raw = (1.0f - acc[r]) * 0.5f;
            float cx  = __expf((1.0f - raw * invM) * INV_SIG) * invS;
            rmax[r] = fmaxf(rmax[r], cx);
        }
    }
    // Reduce over the 16 columns within each lane half (rows live per-half).
    #pragma unroll
    for (int r = 0; r < 8; ++r) {
        #pragma unroll
        for (int msk = 1; msk < 16; msk <<= 1)
            rmax[r] = fmaxf(rmax[r], __shfl_xor(rmax[r], msk, 32));
    }
    __shared__ float s_row[4][16];
    if ((lane & 15) == 0) {
        int half = lane >> 4;                       // half 0: rows m0+r, half 1: m0+8+r
        #pragma unroll
        for (int r = 0; r < 8; ++r) s_row[wave][half * 8 + r] = rmax[r];
    }
    __syncthreads();
    if (threadIdx.x < 16) {
        float m = fmaxf(fmaxf(s_row[0][threadIdx.x], s_row[1][threadIdx.x]),
                        fmaxf(s_row[2][threadIdx.x], s_row[3][threadIdx.x]));
        rowMax[n * P_ + m0 + threadIdx.x] = m;
    }
}

// Kernel 5: loss = mean_n( -log( mean_p(rowMax) + eps ) )
__global__ void cx_finalize(const float* __restrict__ rowMax, float* __restrict__ out) {
    __shared__ float sb[256];
    int t = threadIdx.x;
    float loss = 0.f;
    for (int n = 0; n < N_; ++n) {
        float s = 0.f;
        for (int p = t; p < P_; p += 256) s += rowMax[n * P_ + p];
        sb[t] = s;
        __syncthreads();
        for (int off = 128; off > 0; off >>= 1) {
            if (t < off) sb[t] += sb[t + off];
            __syncthreads();
        }
        if (t == 0) loss += -logf(sb[0] * (1.0f / P_) + EPS_);
        __syncthreads();
    }
    if (t == 0) out[0] = loss * (1.0f / N_);
}

// ---------------------------------------------------------------------------
extern "C" void kernel_launch(void* const* d_in, const int* in_sizes, int n_in,
                              void* d_out, int out_size, void* d_ws, size_t ws_size,
                              hipStream_t stream) {
    const float* featureT = (const float*)d_in[0];
    const float* featureI = (const float*)d_in[1];
    float* out = (float*)d_out;

    // Workspace layout (~17 MB total):
    //   [0, 1KB)      meanT (256 f32)
    //   [+0, +8MB)    fTt  bf16 [n,p,c]
    //   [+8, +16MB)   fIt  bf16 [n,p,c]
    //   then colMin, colS, rowMax (N*P f32 each = 64KB each)
    char* ws = (char*)d_ws;
    const size_t featBytes = (size_t)N_ * P_ * C_ * 2;  // 8 MB each
    float*  meanT  = (float*)ws;
    __bf16* fTt    = (__bf16*)(ws + 1024);
    __bf16* fIt    = (__bf16*)(ws + 1024 + featBytes);
    float*  colMin = (float*)(ws + 1024 + 2 * featBytes);
    float*  colS   = colMin + (size_t)N_ * P_;
    float*  rowMax = colS   + (size_t)N_ * P_;

    cx_mean    <<<C_,            256, 0, stream>>>(featureT, meanT);
    cx_norm    <<<(N_*P_)/256,   256, 0, stream>>>(featureT, meanT, fTt);
    cx_norm    <<<(N_*P_)/256,   256, 0, stream>>>(featureI, meanT, fIt);
    cx_colstats<<<N_*(P_/16),    128, 0, stream>>>(fTt, fIt, colMin, colS);
    cx_rowmax  <<<N_*(P_/16),    128, 0, stream>>>(fTt, fIt, colMin, colS, rowMax);
    cx_finalize<<<1,             256, 0, stream>>>(rowMax, out);
}
